// GaussianMixtureModel_37211596652560
// MI455X (gfx1250) — compile-verified, matrix operand order checked by reading, not run on hardware
//
#include <hip/hip_runtime.h>
#include <math.h>

// ---- CDNA5 WMMA vector types (wave32) ----
typedef __attribute__((ext_vector_type(16))) __bf16 v16bf;
typedef __attribute__((ext_vector_type(8)))  float  v8f;

#define GMM_K 16
#define GMM_D 64
#define LOG_2PI 1.8378770664093453f

// workspace layout (bytes):
//   Mhi : K * 4096 bf16   = 131072
//   Mlo : K * 4096 bf16   = 131072
//   w   : K * 64  f32     =   4096
//   lc  : K       f32     =     64
#define WS_MHI 0
#define WS_MLO (16 * 4096 * 2)
#define WS_W   (2 * 16 * 4096 * 2)
#define WS_LC  (WS_W + 16 * 64 * 4)

static __device__ __forceinline__ unsigned short bf16_bits(__bf16 b) {
    unsigned short u;
    __builtin_memcpy(&u, &b, 2);
    return u;
}

// ---------------------------------------------------------------------------
// Prep: per-component Cholesky + triangular inverse + fragment emission.
// One block (64 threads) per component k.
// ---------------------------------------------------------------------------
__global__ __launch_bounds__(64)
void gmm_prep(const float* __restrict__ covs, const float* __restrict__ mus,
              const float* __restrict__ pi,
              unsigned short* __restrict__ Mhi, unsigned short* __restrict__ Mlo,
              float* __restrict__ wvec, float* __restrict__ lcv)
{
    __shared__ float A[64][65];   // cov -> L (lower, in place)
    __shared__ float Mi[64][65];  // L^{-1} (lower)
    __shared__ float red[64];

    const int k = blockIdx.x;
    const int t = threadIdx.x;
    const float* C = covs + (size_t)k * GMM_D * GMM_D;

    for (int j = 0; j < GMM_D; ++j) A[t][j] = C[t * GMM_D + j];
    __syncthreads();

    // Right-looking Cholesky, thread t owns row t (lower triangle only).
    for (int j = 0; j < GMM_D; ++j) {
        if (t == j) A[j][j] = sqrtf(A[j][j]);
        __syncthreads();
        if (t > j) A[t][j] = A[t][j] / A[j][j];
        __syncthreads();
        if (t > j) {
            float ltj = A[t][j];
            for (int c = j + 1; c <= t; ++c) A[t][c] -= ltj * A[c][j];
        }
        __syncthreads();
    }

    red[t] = logf(A[t][t]);
    __syncthreads();

    // Triangular inverse: thread t solves column t of Mi (forward substitution).
    {
        const int c = t;
        Mi[c][c] = 1.0f / A[c][c];
        for (int i = c + 1; i < GMM_D; ++i) {
            float s = 0.0f;
            for (int j = c; j < i; ++j) s += A[i][j] * Mi[j][c];
            Mi[i][c] = -s / A[i][i];
        }
    }
    __syncthreads();

    // w_k = Mi * mu_k  (row t; Mi lower triangular)
    {
        const float* mu = mus + k * GMM_D;
        float s = 0.0f;
        for (int j = 0; j <= t; ++j) s += Mi[t][j] * mu[j];
        wvec[k * GMM_D + t] = s;
    }
    if (t == 0) {
        float hl = 0.0f;
        for (int i = 0; i < GMM_D; ++i) hl += red[i];
        lcv[k] = logf(pi[k]) - hl - 0.5f * (float)GMM_D * LOG_2PI;
    }

    // Emit WMMA 16x32 bf16 A-fragments (hi + lo split), pre-swizzled into the
    // exact per-lane element order used by v_wmma_f32_16x16x32_bf16:
    //   fragment f = r*2 + kk ; lane ln ; element j
    //   row = 16*r + (ln&15)
    //   col = 32*kk + 16*(j>>3) + 8*(ln>>4) + (j&7)
    // Upper triangle is stored as zero (lower-triangular M); the main kernel
    // additionally skips the all-zero kk=1 chunks for r=0,1.
    for (int e = t; e < 4096; e += 64) {
        int f    = e >> 9;
        int rem  = e & 511;
        int ln   = rem >> 4;
        int j    = rem & 15;
        int r    = f >> 1;
        int kk   = f & 1;
        int half = ln >> 4;
        int row  = 16 * r + (ln & 15);
        int col  = 32 * kk + 16 * (j >> 3) + 8 * half + (j & 7);
        float v  = (col <= row) ? Mi[row][col] : 0.0f;
        __bf16 h = (__bf16)v;
        __bf16 l = (__bf16)(v - (float)h);
        Mhi[(size_t)k * 4096 + e] = bf16_bits(h);
        Mlo[(size_t)k * 4096 + e] = bf16_bits(l);
    }
}

// ---------------------------------------------------------------------------
// Main: one wave per 32 points (two 16-point tiles).
// z^T[64x16] = M_k[64x64] @ x^T[64x16] via bf16x3 WMMA (triangular chunks of
// M skipped at compile time), maha = ||z - w_k||^2, softmax over components.
// ---------------------------------------------------------------------------
__global__ __launch_bounds__(256)
void gmm_main(const float* __restrict__ x,
              const unsigned short* __restrict__ Mhi,
              const unsigned short* __restrict__ Mlo,
              const float* __restrict__ wvec, const float* __restrict__ lcv,
              float* __restrict__ out, int n)
{
    __shared__ float slog[8][32][17];  // [warp][point-in-macro-tile][k], padded

    const int warp = threadIdx.x >> 5;
    const int lane = threadIdx.x & 31;
    const int base = (blockIdx.x * 8 + warp) * 32;
    if (base >= n) return;              // wave-uniform: EXEC stays all-ones
    const int half = lane >> 4;
    const int pr   = lane & 15;

    // ---- Build B fragments (x^T, bf16 hi/lo) for both 16-point tiles ----
    // B element j of lane ln: N = ln&15 (point), K = 16*(j>>3) + 8*half + (j&7)
    v16bf Bh[2][2], Bl[2][2];
    #pragma unroll
    for (int tile = 0; tile < 2; ++tile) {
        const float* xr = x + (size_t)(base + 16 * tile + pr) * GMM_D;
        #pragma unroll
        for (int kk = 0; kk < 2; ++kk) {
            float vals[16];
            #pragma unroll
            for (int g = 0; g < 2; ++g) {
                const float4* p = (const float4*)(xr + 32 * kk + 16 * g + 8 * half);
                float4 a = p[0], b = p[1];
                vals[8 * g + 0] = a.x; vals[8 * g + 1] = a.y;
                vals[8 * g + 2] = a.z; vals[8 * g + 3] = a.w;
                vals[8 * g + 4] = b.x; vals[8 * g + 5] = b.y;
                vals[8 * g + 6] = b.z; vals[8 * g + 7] = b.w;
            }
            v16bf h, l;
            #pragma unroll
            for (int j = 0; j < 16; ++j) {
                __bf16 hv = (__bf16)vals[j];
                h[j] = hv;
                l[j] = (__bf16)(vals[j] - (float)hv);
            }
            Bh[tile][kk] = h;
            Bl[tile][kk] = l;
        }
    }

    // ---- Component loop ----
    for (int k = 0; k < GMM_K; ++k) {
        float lck = lcv[k];
        float ss0 = 0.0f, ss1 = 0.0f;
        #pragma unroll
        for (int r = 0; r < 4; ++r) {
            // M is lower triangular: row-tiles r=0,1 only touch cols 0..31,
            // so the kk=1 contraction chunk is identically zero -> skip it.
            const bool need_kk1 = (r >= 2);
            const size_t fb = (size_t)k * 4096 + (size_t)(r * 2) * 512 + (size_t)lane * 16;
            v16bf Ah0 = *(const v16bf*)(Mhi + fb);
            v16bf Al0 = *(const v16bf*)(Mlo + fb);
            v16bf Ah1 = {}, Al1 = {};
            if (need_kk1) {
                Ah1 = *(const v16bf*)(Mhi + fb + 512);
                Al1 = *(const v16bf*)(Mlo + fb + 512);
            }
            const float4* wp = (const float4*)(wvec + k * GMM_D + 16 * r + 8 * half);
            float4 w0 = wp[0], w1 = wp[1];
            #pragma unroll
            for (int tile = 0; tile < 2; ++tile) {
                v8f acc = {0.f, 0.f, 0.f, 0.f, 0.f, 0.f, 0.f, 0.f};
                // bf16x3: hi*hi + hi*lo + lo*hi  (f32 accumulate)
                acc = __builtin_amdgcn_wmma_f32_16x16x32_bf16(false, Ah0, false, Bh[tile][0], (short)0, acc, false, false);
                acc = __builtin_amdgcn_wmma_f32_16x16x32_bf16(false, Ah0, false, Bl[tile][0], (short)0, acc, false, false);
                acc = __builtin_amdgcn_wmma_f32_16x16x32_bf16(false, Al0, false, Bh[tile][0], (short)0, acc, false, false);
                if (need_kk1) {
                    acc = __builtin_amdgcn_wmma_f32_16x16x32_bf16(false, Ah1, false, Bh[tile][1], (short)0, acc, false, false);
                    acc = __builtin_amdgcn_wmma_f32_16x16x32_bf16(false, Ah1, false, Bl[tile][1], (short)0, acc, false, false);
                    acc = __builtin_amdgcn_wmma_f32_16x16x32_bf16(false, Al1, false, Bh[tile][1], (short)0, acc, false, false);
                }
                // z = (M x) - w ; accumulate z^2 over the 8 rows this lane holds
                float d0 = acc[0] - w0.x, d1 = acc[1] - w0.y;
                float d2 = acc[2] - w0.z, d3 = acc[3] - w0.w;
                float d4 = acc[4] - w1.x, d5 = acc[5] - w1.y;
                float d6 = acc[6] - w1.z, d7 = acc[7] - w1.w;
                float zs = d0*d0 + d1*d1 + d2*d2 + d3*d3 + d4*d4 + d5*d5 + d6*d6 + d7*d7;
                if (tile == 0) ss0 += zs; else ss1 += zs;
            }
        }
        // combine the two row-halves (lanes l and l^16 hold same point, disjoint rows)
        float m0 = ss0 + __shfl_xor(ss0, 16, 32);
        float m1 = ss1 + __shfl_xor(ss1, 16, 32);
        float lp = -0.5f * (half ? m1 : m0) + lck;
        slog[warp][half * 16 + pr][k] = lp;   // lane's tile == its half
    }

    // ---- softmax over k for this lane's point (LDS RAW same-wave: in order) ----
    float v[16];
    #pragma unroll
    for (int k = 0; k < GMM_K; ++k) v[k] = slog[warp][half * 16 + pr][k];
    float mx = v[0];
    #pragma unroll
    for (int k = 1; k < GMM_K; ++k) mx = fmaxf(mx, v[k]);
    float e[16];
    float s = 0.0f;
    #pragma unroll
    for (int k = 0; k < GMM_K; ++k) { e[k] = expf(v[k] - mx); s += e[k]; }
    float inv = 1.0f / s;

    const int point = base + half * 16 + pr;
    float4* op = (float4*)(out + (size_t)point * GMM_K);
    op[0] = make_float4(e[0] * inv,  e[1] * inv,  e[2] * inv,  e[3] * inv);
    op[1] = make_float4(e[4] * inv,  e[5] * inv,  e[6] * inv,  e[7] * inv);
    op[2] = make_float4(e[8] * inv,  e[9] * inv,  e[10] * inv, e[11] * inv);
    op[3] = make_float4(e[12] * inv, e[13] * inv, e[14] * inv, e[15] * inv);
}

// ---------------------------------------------------------------------------
extern "C" void kernel_launch(void* const* d_in, const int* in_sizes, int n_in,
                              void* d_out, int out_size, void* d_ws, size_t ws_size,
                              hipStream_t stream) {
    const float* x    = (const float*)d_in[0];
    const float* mus  = (const float*)d_in[1];
    const float* covs = (const float*)d_in[2];
    const float* pi   = (const float*)d_in[3];
    float* out = (float*)d_out;

    char* ws = (char*)d_ws;
    unsigned short* Mhi = (unsigned short*)(ws + WS_MHI);
    unsigned short* Mlo = (unsigned short*)(ws + WS_MLO);
    float*          wv  = (float*)(ws + WS_W);
    float*          lc  = (float*)(ws + WS_LC);

    const int n = in_sizes[0] / GMM_D;  // N points

    gmm_prep<<<GMM_K, 64, 0, stream>>>(covs, mus, pi, Mhi, Mlo, wv, lc);

    const int waves  = (n + 31) / 32;       // one wave per 32 points
    const int blocks = (waves + 7) / 8;     // 8 waves (256 threads) per block
    gmm_main<<<blocks, 256, 0, stream>>>(x, Mhi, Mlo, wv, lc, out, n);
}
